// BERT_CRF_NER_49993419325864
// MI455X (gfx1250) — matrix operand order repaired and checked by reading, not compiled, not run on hardware
//
#include <hip/hip_runtime.h>
#include <hip/hip_bf16.h>

typedef __attribute__((ext_vector_type(16))) _Float16 v16h;
typedef __attribute__((ext_vector_type(8)))  _Float16 v8h;
typedef __attribute__((ext_vector_type(8)))  float    v8f;

#define B_    128
#define T_    512
#define H_    768
#define NPAD  80          // padded label columns: dis[0:6), ner[16:29), pos[32:80)
#define NEG_INF_F (-10000.0f)

// ---------------- workspace layout (bytes) ----------------
// [0, 122880)                 W_pack : 80 x 768 f16 (pre-converted)
// [131072, 131392)            b_pack : 80 f32
// [262144, 262144+20971520)   feats  : 65536 x 80 f32
// [+0, +1536)                 part   : 3 x 128 f32 (per-task per-batch loss)
#define WS_WPACK   0
#define WS_BPACK   131072
#define WS_FEATS   262144
#define WS_PART    (262144 + 65536 * 80 * 4)

// ---------------------------------------------------------------------------
// Pack the three weight matrices + biases into 16-aligned column tiles,
// converting weights to f16 ONCE (removes per-block cvt work in the GEMM).
// ---------------------------------------------------------------------------
__global__ void pack_weights(const float* __restrict__ Wd, const float* __restrict__ bd,
                             const float* __restrict__ Wn, const float* __restrict__ bn,
                             const float* __restrict__ Wp, const float* __restrict__ bp,
                             _Float16* __restrict__ Wpack, float* __restrict__ bpack) {
    int idx = blockIdx.x * blockDim.x + threadIdx.x;
    if (idx < NPAD * H_) {
        int n = idx / H_, k = idx % H_;
        float w = 0.0f;
        if (n < 6)                   w = Wd[n * H_ + k];
        else if (n >= 16 && n < 29)  w = Wn[(n - 16) * H_ + k];
        else if (n >= 32 && n < 80)  w = Wp[(n - 32) * H_ + k];
        Wpack[idx] = (_Float16)w;
    }
    if (idx < NPAD) {
        float bias = 0.0f;
        if (idx < 6)                     bias = bd[idx];
        else if (idx >= 16 && idx < 29)  bias = bn[idx - 16];
        else if (idx >= 32)              bias = bp[idx - 32];
        bpack[idx] = bias;
    }
}

// ---------------------------------------------------------------------------
// feats = X (65536x768) * Wpack^T (80x768) + b, via v_wmma_f32_16x16x32_f16.
// One wave per 16-row tile; each wave computes all 5 column tiles so X is
// read exactly once from HBM (memory-bound: 201 MB @ 23.3 TB/s ~ 9 us).
// Weights are L2-resident f16 — B fragments load directly, no cvt.
// ---------------------------------------------------------------------------
__global__ void __launch_bounds__(32)
gemm_feats(const float* __restrict__ X, const _Float16* __restrict__ Wpack,
           const float* __restrict__ bpack, float* __restrict__ feats) {
    const int lane = threadIdx.x;            // 0..31 (wave32)
    const int row0 = blockIdx.x * 16;        // M tile
    const int n    = lane & 15;
    const int hi   = lane >> 4;              // 0 or 1 (lane half)

    v8f acc[5] = {};                         // 5 N-tiles of 16x16 f32

    // A-fragment row for this lane (16-bit A 16x32 layout, ISA 7.12.2)
    const float* Xrow = X + (size_t)(row0 + n) * H_;

    for (int kt = 0; kt < H_; kt += 32) {
        if (kt + 32 < H_)
            __builtin_prefetch(Xrow + kt + 32 + 8 * hi, 0, 1);  // global_prefetch_b8

        // ---- A fragment: lane holds K = kt + 16*g + 8*hi + [0..8) for g=0,1
        v16h a;
#pragma unroll
        for (int g = 0; g < 2; ++g) {
            const float4* p = reinterpret_cast<const float4*>(Xrow + kt + 16 * g + 8 * hi);
            float4 c0 = p[0], c1 = p[1];
            a[8 * g + 0] = (_Float16)c0.x; a[8 * g + 1] = (_Float16)c0.y;
            a[8 * g + 2] = (_Float16)c0.z; a[8 * g + 3] = (_Float16)c0.w;
            a[8 * g + 4] = (_Float16)c1.x; a[8 * g + 5] = (_Float16)c1.y;
            a[8 * g + 6] = (_Float16)c1.z; a[8 * g + 7] = (_Float16)c1.w;
        }

        // ---- 5 B fragments (f16, L2-resident), one WMMA each
#pragma unroll
        for (int nt = 0; nt < 5; ++nt) {
            const _Float16* Wrow = Wpack + (size_t)(nt * 16 + n) * H_ + kt + 8 * hi;
            const v8h* wp = reinterpret_cast<const v8h*>(Wrow);
            v8h w0 = wp[0], w1 = wp[2];
            v16h b;
#pragma unroll
            for (int j = 0; j < 8; ++j) { b[j] = w0[j]; b[8 + j] = w1[j]; }
            acc[nt] = __builtin_amdgcn_wmma_f32_16x16x32_f16(
                false, a, false, b, (short)0, acc[nt], false, false);
        }
    }

    // Epilogue: C/D layout — VGPR r: lanes0-15 M=r, lanes16-31 M=r+8, N=lane%16
#pragma unroll
    for (int nt = 0; nt < 5; ++nt) {
        float bias = bpack[nt * 16 + n];
#pragma unroll
        for (int r = 0; r < 8; ++r) {
            int mr = r + 8 * hi;
            feats[(size_t)(row0 + mr) * NPAD + nt * 16 + n] = acc[nt][r] + bias;
        }
    }
}

// ---------------------------------------------------------------------------
// Per (batch, task): CRF forward log-partition (511-step LSE scan, alpha in
// LDS double buffer) minus gold path score (fused, tree-reduced).
// ---------------------------------------------------------------------------
__global__ void __launch_bounds__(64)
crf_loss(const float* __restrict__ feats,
         const float* __restrict__ trans_dis, const float* __restrict__ trans_ner,
         const float* __restrict__ trans_pos,
         const int* __restrict__ y_dis, const int* __restrict__ y_ner,
         const int* __restrict__ y_pos,
         float* __restrict__ part) {
    __shared__ float trans_s[48 * 48];
    __shared__ float la[2][48];
    __shared__ float red[64];

    const int b    = blockIdx.x;
    const int task = blockIdx.y;
    const int tid  = threadIdx.x;

    int L, off, start;
    const float* trans;
    const int*   y;
    if (task == 0)      { L = 6;  off = 0;  start = 4;  trans = trans_dis; y = y_dis; }
    else if (task == 1) { L = 13; off = 16; start = 11; trans = trans_ner; y = y_ner; }
    else                { L = 48; off = 32; start = 46; trans = trans_pos; y = y_pos; }

    for (int i = tid; i < L * L; i += 64) trans_s[i] = trans[i];
    if (tid < 48) la[0][tid] = (tid == start) ? 0.0f : NEG_INF_F;
    __syncthreads();

    // ---- gold path score (t = 1..511), parallel over t
    float g = 0.0f;
    for (int t = 1 + tid; t < T_; t += 64) {
        int yt = y[b * T_ + t];
        int yp = y[b * T_ + t - 1];
        g += trans_s[yt * L + yp] + feats[(size_t)(b * T_ + t) * NPAD + off + yt];
    }
    red[tid] = g;
    __syncthreads();
    for (int s2 = 32; s2 > 0; s2 >>= 1) {
        if (tid < s2) red[tid] += red[tid + s2];
        __syncthreads();
    }
    float gold = red[0];
    __syncthreads();

    // ---- forward scan: la'[l] = LSE_{l'}(trans[l,l'] + la[l']) + feat[t,l]
    const float* fb = feats + (size_t)b * T_ * NPAD + off;
    int cur = 0;
    for (int t = 1; t < T_; ++t) {
        if (tid < L) {
            const float* tr = trans_s + tid * L;
            float m = -3.0e38f;
            for (int lp = 0; lp < L; ++lp) {
                float v = tr[lp] + la[cur][lp];
                m = fmaxf(m, v);
            }
            float s = 0.0f;
            for (int lp = 0; lp < L; ++lp)
                s += __expf(tr[lp] + la[cur][lp] - m);
            la[cur ^ 1][tid] = m + __logf(s) + fb[(size_t)t * NPAD + tid];
        }
        __syncthreads();
        cur ^= 1;
    }

    if (tid == 0) {
        float m = -3.0e38f;
        for (int l = 0; l < L; ++l) m = fmaxf(m, la[cur][l]);
        float s = 0.0f;
        for (int l = 0; l < L; ++l) s += __expf(la[cur][l] - m);
        part[task * B_ + b] = (m + __logf(s)) - gold;
    }
}

// ---------------------------------------------------------------------------
// loss = mean_b( dis + 0.5*ner + 0.5*pos )
// ---------------------------------------------------------------------------
__global__ void __launch_bounds__(128)
final_reduce(const float* __restrict__ part, float* __restrict__ out) {
    __shared__ float red[128];
    int b = threadIdx.x;
    red[b] = part[b] + 0.5f * part[B_ + b] + 0.5f * part[2 * B_ + b];
    __syncthreads();
    for (int s = 64; s > 0; s >>= 1) {
        if (b < s) red[b] += red[b + s];
        __syncthreads();
    }
    if (b == 0) out[0] = red[0] / (float)B_;
}

extern "C" void kernel_launch(void* const* d_in, const int* in_sizes, int n_in,
                              void* d_out, int out_size, void* d_ws, size_t ws_size,
                              hipStream_t stream) {
    (void)in_sizes; (void)n_in; (void)out_size; (void)ws_size;

    const float* X     = (const float*)d_in[0];
    const int*   y_dis = (const int*)d_in[1];
    const int*   y_ner = (const int*)d_in[2];
    const int*   y_pos = (const int*)d_in[3];
    const float* Wd    = (const float*)d_in[4];
    const float* bd    = (const float*)d_in[5];
    const float* Wn    = (const float*)d_in[6];
    const float* bn    = (const float*)d_in[7];
    const float* Wp    = (const float*)d_in[8];
    const float* bp    = (const float*)d_in[9];
    const float* td    = (const float*)d_in[10];
    const float* tn    = (const float*)d_in[11];
    const float* tp    = (const float*)d_in[12];

    char*     ws    = (char*)d_ws;
    _Float16* Wpack = (_Float16*)(ws + WS_WPACK);
    float*    bpack = (float*)(ws + WS_BPACK);
    float*    feats = (float*)(ws + WS_FEATS);
    float*    part  = (float*)(ws + WS_PART);

    pack_weights<<<(NPAD * H_ + 255) / 256, 256, 0, stream>>>(
        Wd, bd, Wn, bn, Wp, bp, Wpack, bpack);

    gemm_feats<<<(B_ * T_) / 16, 32, 0, stream>>>(X, Wpack, bpack, feats);

    crf_loss<<<dim3(B_, 3), 64, 0, stream>>>(
        feats, td, tn, tp, y_dis, y_ner, y_pos, part);

    final_reduce<<<1, 128, 0, stream>>>(part, (float*)d_out);
}
